// ScaledDotProduct_13649406067531
// MI455X (gfx1250) — matrix operand order repaired and checked by reading, not compiled, not run on hardware
//
#include <hip/hip_runtime.h>

typedef __attribute__((ext_vector_type(16))) _Float16 v16h;
typedef __attribute__((ext_vector_type(8)))  _Float16 v8h;
typedef __attribute__((ext_vector_type(8)))  float    v8f;
typedef __attribute__((ext_vector_type(4)))  float    vf4;

#define WAVES 8
#define OB_STRIDE 68   // floats per O-row in LDS transpose buffer (272B, 16B-aligned)
#define KS_STRIDE 72   // halves per staged K row (144B: 16B-aligned, bank-permuting)
#define VS_STRIDE 40   // halves per staged V^T row (80B: 16B-aligned, bank-permuting)

#if __has_builtin(__builtin_amdgcn_exp2f)
#define EXP2F(x) __builtin_amdgcn_exp2f(x)
#else
#define EXP2F(x) exp2f(x)
#endif

#define CAT16(lo, hi) __builtin_shufflevector(lo, hi, 0,1,2,3,4,5,6,7,8,9,10,11,12,13,14,15)

union SMem {
    struct Stage {
        _Float16 k[32 * KS_STRIDE];   // K tile, row-major [j_local][d]
        _Float16 v[64 * VS_STRIDE];   // V tile, transposed [d][j_local]
    } st;
    float ob[WAVES][16 * OB_STRIDE];  // end-of-kernel O^T -> O transpose (reused)
};

__global__ __launch_bounds__(32 * WAVES)
void fa_causal_wmma_kernel(const float* __restrict__ Kg,
                           const float* __restrict__ Qg,
                           const float* __restrict__ Vg,
                           float* __restrict__ Og)
{
    constexpr int S = 2048, D = 64;
    constexpr int ROWS_PER_BLOCK = 16 * WAVES;          // 128 q-rows per workgroup
    constexpr int QBLOCKS = S / ROWS_PER_BLOCK;         // 16

    __shared__ __align__(16) SMem smem;

    const int tid  = threadIdx.x;
    const int wave = tid >> 5;
    const int lane = tid & 31;
    const int half = lane >> 4;     // 0: lanes 0-15, 1: lanes 16-31
    const int n    = lane & 15;

    const int bh = blockIdx.x / QBLOCKS;
    const int qb = blockIdx.x % QBLOCKS;
    const int qB = qb * ROWS_PER_BLOCK;    // workgroup's first q-row
    const int q0 = qB + wave * 16;         // this wave's q-tile

    const float* Qp = Qg + (size_t)bh * S * D;
    const float* Kp = Kg + (size_t)bh * S * D;
    const float* Vp = Vg + (size_t)bh * S * D;
    float*       Op = Og + (size_t)bh * S * D;

    // ---- Q as B-fragment of Q^T: col q=q0+n, contraction d = 32kc + 16half + i.
    // Fold 1/sqrt(D) and log2(e) (exp2 used throughout).
    const float qscale = 0.125f * 1.44269504f;
    v16h qfrag[2];
#pragma unroll
    for (int kc = 0; kc < 2; ++kc) {
        const float* rp = Qp + (size_t)(q0 + n) * D + kc * 32 + half * 16;
#pragma unroll
        for (int i = 0; i < 16; i += 4) {
            vf4 x = *(const vf4*)(rp + i);
            qfrag[kc][i]     = (_Float16)(x[0] * qscale);
            qfrag[kc][i + 1] = (_Float16)(x[1] * qscale);
            qfrag[kc][i + 2] = (_Float16)(x[2] * qscale);
            qfrag[kc][i + 3] = (_Float16)(x[3] * qscale);
        }
    }

    float m = -3.0e38f, l = 0.0f;
    v8f oacc[4] = {};   // O^T: VGPR r -> d = 16c + r + 8half, col q = n

    // cooperative-staging thread mapping: 256 threads cover a 32x64 tile, 8 floats each
    const int srow = tid >> 3;        // 0..31 (j_local)
    const int sd0  = (tid & 7) * 8;   // 0,8,..,56 (d)

    const int jmax  = qB + ROWS_PER_BLOCK - 1;   // workgroup-uniform causal extent
    const int qLast = q0 + 15;                   // this wave's causal extent

    for (int j0 = 0; j0 <= jmax; j0 += 32) {
        // ---- cooperative staging: global loads first (overlap latency with barrier)
        const float* krp = Kp + (size_t)(j0 + srow) * D + sd0;
        vf4 ka0 = *(const vf4*)(krp);
        vf4 ka1 = *(const vf4*)(krp + 4);
        const float* vrp = Vp + (size_t)(j0 + srow) * D + sd0;
        vf4 va0 = *(const vf4*)(vrp);
        vf4 va1 = *(const vf4*)(vrp + 4);

        __syncthreads();   // previous tile fully consumed

        v8h kh;
#pragma unroll
        for (int i = 0; i < 4; ++i) { kh[i] = (_Float16)ka0[i]; kh[4 + i] = (_Float16)ka1[i]; }
        *(v8h*)(&smem.st.k[srow * KS_STRIDE + sd0]) = kh;

        smem.st.v[(sd0 + 0) * VS_STRIDE + srow] = (_Float16)va0[0];
        smem.st.v[(sd0 + 1) * VS_STRIDE + srow] = (_Float16)va0[1];
        smem.st.v[(sd0 + 2) * VS_STRIDE + srow] = (_Float16)va0[2];
        smem.st.v[(sd0 + 3) * VS_STRIDE + srow] = (_Float16)va0[3];
        smem.st.v[(sd0 + 4) * VS_STRIDE + srow] = (_Float16)va1[0];
        smem.st.v[(sd0 + 5) * VS_STRIDE + srow] = (_Float16)va1[1];
        smem.st.v[(sd0 + 6) * VS_STRIDE + srow] = (_Float16)va1[2];
        smem.st.v[(sd0 + 7) * VS_STRIDE + srow] = (_Float16)va1[3];

        __syncthreads();   // tile ready

        if (j0 <= qLast) {   // wave-uniform: EXEC stays all-ones for WMMA
            // ---- S^T = K Q^T : A = K tile from LDS
            v8f sfr[2];
#pragma unroll
            for (int t = 0; t < 2; ++t) {
                const int krl = t * 16 + n;
                v8f acc = {};
#pragma unroll
                for (int kc = 0; kc < 2; ++kc) {
                    const _Float16* rp = &smem.st.k[krl * KS_STRIDE + kc * 32 + half * 16];
                    v8h lo = *(const v8h*)(rp);        // d = 32kc + 16half + 0..7
                    v8h hi = *(const v8h*)(rp + 8);    // d = 32kc + 16half + 8 + 0..7
                    v16h ka = CAT16(lo, hi);
                    acc = __builtin_amdgcn_wmma_f32_16x16x32_f16(
                        false, ka, false, qfrag[kc], (short)0, acc, false, false);
                }
                sfr[t] = acc;
            }

            // ---- causal mask (diagonal blocks only)
            if (j0 + 31 > q0) {
#pragma unroll
                for (int t = 0; t < 2; ++t)
#pragma unroll
                    for (int r = 0; r < 8; ++r)
                        if (j0 + t * 16 + r + 8 * half > q0 + n) sfr[t][r] = -1.0e30f;
            }

            // ---- online softmax: in-lane reduce + one half-wave swap
            float mx = fmaxf(sfr[0][0], sfr[1][0]);
#pragma unroll
            for (int r = 1; r < 8; ++r) mx = fmaxf(mx, fmaxf(sfr[0][r], sfr[1][r]));
            mx = fmaxf(mx, __shfl_xor(mx, 16, 32));
            const float mnew = fmaxf(m, mx);
            const float alpha = EXP2F(m - mnew);
            m = mnew;
            float rs = 0.0f;
#pragma unroll
            for (int t = 0; t < 2; ++t)
#pragma unroll
                for (int r = 0; r < 8; ++r) {
                    const float p = EXP2F(sfr[t][r] - mnew);
                    sfr[t][r] = p;
                    rs += p;
                }
            rs += __shfl_xor(rs, 16, 32);
            l = l * alpha + rs;
#pragma unroll
            for (int c = 0; c < 4; ++c)
#pragma unroll
                for (int r = 0; r < 8; ++r) oacc[c][r] *= alpha;

            // ---- P^T B-fragment: half-wave j-swap
            v16h pb;
#pragma unroll
            for (int r = 0; r < 8; ++r) {
                const float snd = half ? sfr[0][r] : sfr[1][r];
                const float rcv = __shfl_xor(snd, 16, 32);
                const float lo = half ? rcv : sfr[0][r];      // j = 16half + r
                const float hi = half ? sfr[1][r] : rcv;      // j = 16half + 8 + r
                pb[r]     = (_Float16)lo;
                pb[8 + r] = (_Float16)hi;
            }

            // ---- O^T += V^T P^T : A = V^T tile rows from LDS
#pragma unroll
            for (int c = 0; c < 4; ++c) {
                const _Float16* rp = &smem.st.v[(c * 16 + n) * VS_STRIDE + half * 16];
                v8h lo = *(const v8h*)(rp);        // j_local = 16half + 0..7
                v8h hi = *(const v8h*)(rp + 8);    // j_local = 16half + 8 + 0..7
                v16h va = CAT16(lo, hi);
                oacc[c] = __builtin_amdgcn_wmma_f32_16x16x32_f16(
                    false, va, false, pb, (short)0, oacc[c], false, false);
            }
        }
    }

    __syncthreads();   // all waves done with staging buffers -> safe to reuse as obuf

    // ---- normalize (per-lane scalar), transpose O^T -> O via LDS, coalesced store
    const float linv = 1.0f / l;
    float* ob = smem.ob[wave];
#pragma unroll
    for (int c = 0; c < 4; ++c)
#pragma unroll
        for (int r = 0; r < 8; ++r)
            ob[n * OB_STRIDE + c * 16 + r + 8 * half] = oacc[c][r] * linv;

    // same-wave DS ops are in-order: safe to read back without a barrier
    const int row = lane >> 1;
    const int cb  = (lane & 1) * 32;
#pragma unroll
    for (int i = 0; i < 8; ++i) {
        vf4 x = *(const vf4*)(ob + row * OB_STRIDE + cb + i * 4);
        *(vf4*)(Op + (size_t)(q0 + row) * D + cb + i * 4) = x;
    }
}

extern "C" void kernel_launch(void* const* d_in, const int* in_sizes, int n_in,
                              void* d_out, int out_size, void* d_ws, size_t ws_size,
                              hipStream_t stream) {
    // setup_inputs order: keys, queries, values (all fp32 [B,H,S,D])
    const float* Kp = (const float*)d_in[0];
    const float* Qp = (const float*)d_in[1];
    const float* Vp = (const float*)d_in[2];
    float* Op = (float*)d_out;

    constexpr int B = 4, H = 12, S = 2048;
    constexpr int QBLOCKS = S / (16 * WAVES);       // 16 workgroups of 128 q-rows per bh
    dim3 grid(B * H * QBLOCKS);                     // 768 blocks
    dim3 block(32 * WAVES);                         // 256 threads = 8 wave32
    fa_causal_wmma_kernel<<<grid, block, 0, stream>>>(Kp, Qp, Vp, Op);
}
